// NgramMultiheadAttention_61392262529340
// MI455X (gfx1250) — compile-verified, matrix operand order checked by reading, not run on hardware
//
#include <hip/hip_runtime.h>
#include <hip/hip_bf16.h>

typedef __attribute__((ext_vector_type(16))) __bf16 v16bf;
typedef __attribute__((ext_vector_type(8)))  float  v8f;

#define T_DIM 1536
#define RT    512
#define B_DIM 4
#define H_DIM 16
#define HD    64
#define E_DIM 1024
#define M_ROWS (T_DIM * B_DIM)   // 6144

// ---------------- WMMA helpers (CDNA5 16x16x32 bf16 -> f32) ----------------

static __device__ __forceinline__ v8f wmma_bf16(v16bf a, v16bf b, v8f c) {
  return __builtin_amdgcn_wmma_f32_16x16x32_bf16(
      /*neg_a=*/false, a, /*neg_b=*/false, b,
      /*c_mod=*/(short)0, c, /*reuse_a=*/false, /*reuse_b=*/false);
}

// Fragment loader for sources where the WMMA K-dim is contiguous in memory.
//  - A-frag (16 rows x 32 K): lane&15 picks the row (M).
//  - B-frag (32 K x 16 cols) from an (N x K) row-major source: lane&15 picks
//    the column (N).
// Lanes 0-15 hold K {0..7,16..23}+k0, lanes 16-31 hold K {8..15,24..31}+k0.
// Two contiguous 16-byte loads per lane.
static __device__ __forceinline__ v16bf load_frag_rowK(const __bf16* __restrict__ src,
                                                       int ld, int row, int k0) {
  int lane = threadIdx.x & 31;
  int half = lane >> 4;
  int r    = lane & 15;
  const __bf16* p = src + (long)(row + r) * ld + k0 + half * 8;
  union { uint4 u[2]; v16bf f; } t;
  t.u[0] = *(const uint4*)(p);
  t.u[1] = *(const uint4*)(p + 16);
  return t.f;
}

// ---------------- fp32 -> bf16 conversion ----------------

__global__ void cvt_f32_bf16(const float* __restrict__ src, __bf16* __restrict__ dst, long n) {
  long i = (long)blockIdx.x * blockDim.x + threadIdx.x;
  if (i < n) dst[i] = (__bf16)src[i];
}

// ---------------- GEMM: out[m][n] = sum_k A[m][k] * W[n][k] + bias[n] ----------------
// mode 0: write fp32 outF (row-major M x N)
// mode 1: QKV epilogue -> q ([bh][p][hd], scaled), k ([bh][p][hd]), vT ([bh][hd][p]), bf16
// Wave tile: 32x64 (spill-free: 64 acc VGPRs + 2 A frags + 1 B frag).  4 waves/block.

__global__ void __launch_bounds__(128, 1)
gemm_ws(const __bf16* __restrict__ A, const __bf16* __restrict__ W,
        const float* __restrict__ bias, float* __restrict__ outF,
        __bf16* __restrict__ qb, __bf16* __restrict__ kb,
        __bf16* __restrict__ vT, int M, int N, int K, int mode) {
  // wave index made explicitly scalar so tile math / epilogue branches are uniform
  int wave = __builtin_amdgcn_readfirstlane(threadIdx.x >> 5);
  int lane = threadIdx.x & 31;
  int half = lane >> 4;
  int l15  = lane & 15;

  int ntiles_n = N >> 6;
  int tile = blockIdx.x * 4 + wave;
  int mb = tile / ntiles_n;
  int nb = tile % ntiles_n;
  int m0 = mb * 32;
  int n0 = nb * 64;

  v8f acc[2][4];
#pragma unroll
  for (int i = 0; i < 2; ++i)
#pragma unroll
    for (int j = 0; j < 4; ++j)
#pragma unroll
      for (int r = 0; r < 8; ++r) acc[i][j][r] = 0.f;

#pragma unroll 1
  for (int k0 = 0; k0 < K; k0 += 32) {
    __builtin_prefetch(A + (long)m0 * K + k0 + 128, 0, 1);
    __builtin_prefetch(W + (long)n0 * K + k0 + 128, 0, 1);
    v16bf a0 = load_frag_rowK(A, K, m0, k0);
    v16bf a1 = load_frag_rowK(A, K, m0 + 16, k0);
#pragma unroll
    for (int j = 0; j < 4; ++j) {
      v16bf b = load_frag_rowK(W, K, n0 + j * 16, k0);
      acc[0][j] = wmma_bf16(a0, b, acc[0][j]);
      acc[1][j] = wmma_bf16(a1, b, acc[1][j]);
    }
  }

#pragma unroll
  for (int i = 0; i < 2; ++i) {
#pragma unroll
    for (int j = 0; j < 4; ++j) {
      int nbase = n0 + j * 16;          // scalar: uniform across the 16 output columns
      if (mode == 0) {
#pragma unroll
        for (int r = 0; r < 8; ++r) {
          int m = m0 + i * 16 + r + half * 8;
          int n = nbase + l15;
          outF[(long)m * N + n] = acc[i][j][r] + bias[n];
        }
      } else {
        int region = nbase >> 10;       // scalar: 0 = Q, 1 = K, 2 = V
        int h  = (nbase >> 6) & 15;     // scalar head index
        int c0 = nbase & 63;            // scalar channel-tile base within head
        if (region == 0) {
#pragma unroll
          for (int r = 0; r < 8; ++r) {
            int m = m0 + i * 16 + r + half * 8;
            int p = m >> 2, bb = m & 3, bh = bb * H_DIM + h;
            float vv = acc[i][j][r] + bias[nbase + l15];
            qb[((long)bh * T_DIM + p) * HD + c0 + l15] = (__bf16)(vv * 0.125f);
          }
        } else if (region == 1) {
#pragma unroll
          for (int r = 0; r < 8; ++r) {
            int m = m0 + i * 16 + r + half * 8;
            int p = m >> 2, bb = m & 3, bh = bb * H_DIM + h;
            float vv = acc[i][j][r] + bias[nbase + l15];
            kb[((long)bh * T_DIM + p) * HD + c0 + l15] = (__bf16)vv;
          }
        } else {
#pragma unroll
          for (int r = 0; r < 8; ++r) {
            int m = m0 + i * 16 + r + half * 8;
            int p = m >> 2, bb = m & 3, bh = bb * H_DIM + h;
            float vv = acc[i][j][r] + bias[nbase + l15];
            vT[((long)bh * HD + c0 + l15) * T_DIM + p] = (__bf16)vv;
          }
        }
      }
    }
  }
}

// ---------------- Attention (main + ngram unified over global position) -------------
// grid: (BH=64, T/64=24), block 128 (4 waves -> fills all 4 SIMD32s of a WGP),
// each wave: 16 query rows.  LDS: 64KB per wave (scores f32, softmaxed P is
// converted to bf16 IN PLACE over the low half of each score row).

__global__ void __launch_bounds__(128, 1)
attn_kernel(const __bf16* __restrict__ qb, const __bf16* __restrict__ kb,
            const __bf16* __restrict__ vT, const float* __restrict__ vals,
            const float* __restrict__ self_mask,
            const float* __restrict__ ngram_mask,
            const int* __restrict__ ib_main, const int* __restrict__ ib_rel,
            __bf16* __restrict__ Ao) {
  extern __shared__ char smem[];
  int wave = __builtin_amdgcn_readfirstlane(threadIdx.x >> 5);
  int lane = threadIdx.x & 31;
  int half = lane >> 4;
  int l15  = lane & 15;

  int bh = blockIdx.x;
  int b  = bh >> 4;
  int h  = bh & 15;
  int p0 = blockIdx.y * 64 + wave * 16;   // global query row block
  int seg = p0 / RT;                      // 0 = main, 1 = ngram0, 2 = ngram1
  int t0  = p0 % RT;
  int S   = (seg == 0) ? RT : 2 * RT;

  char*   base = smem + wave * (16 * 1024 * 4);
  float*  sc   = (float*)base;            // [16][1024] scores (f32)
  __bf16* pbb  = (__bf16*)base;           // overlay: [16][2048] slots, row stride 2048

  // Q fragments for this 16-row tile (K = hd = 64 -> two 16x32 frags)
  const __bf16* qbase = qb + (long)bh * T_DIM * HD;
  v16bf qa0 = load_frag_rowK(qbase, HD, p0, 0);
  v16bf qa1 = load_frag_rowK(qbase, HD, p0, 32);

  const __bf16* kbase = kb + (long)bh * T_DIM * HD;

  // ---- Phase A: scores = Q K^T + rel_bias + mask -> LDS
#pragma unroll 1
  for (int st = 0; st < S; st += 16) {
    int kp0 = (st < RT) ? st : st + RT * (seg - 1);   // key position of this tile
    v8f acc;
#pragma unroll
    for (int r = 0; r < 8; ++r) acc[r] = 0.f;
    // B-frag: rows = key positions (lane), K = channels (contiguous)
    acc = wmma_bf16(qa0, load_frag_rowK(kbase, HD, kp0, 0), acc);
    acc = wmma_bf16(qa1, load_frag_rowK(kbase, HD, kp0, 32), acc);

#pragma unroll
    for (int r = 0; r < 8; ++r) {
      int mloc = r + half * 8;
      int t = t0 + mloc;
      int s = st + l15;
      long prow = (long)(p0 + mloc) * B_DIM + b;      // query row in vals
      float bias_v, msk;
      if (seg == 0) {
        int idx = ib_main[((long)b * RT + t) * RT + s];
        bias_v  = vals[prow * 512 + idx * H_DIM + h];
        msk     = self_mask[(long)t * RT + s];
      } else {
        int n   = seg - 1;
        int idx = ib_rel[((long)b * RT + t) * (2 * RT) + s];
        bias_v  = vals[prow * 512 + idx * H_DIM + h];
        msk     = ngram_mask[((long)n * RT + t) * (2 * RT) + s];
      }
      sc[mloc * 1024 + s] = acc[r] + bias_v + msk;
    }
  }
  __syncthreads();

  // ---- Phase B: softmax per row; final normalized P converted to bf16 in place.
  // In the ascending strided sweep, iteration k's bf16 stores at byte 2s only
  // clobber f32 indices <= s/2, all fully consumed in earlier iterations (LDS
  // ops from one wave are in order), so the overlay is race-free.
#pragma unroll 1
  for (int r = 0; r < 16; ++r) {
    float mx = -3.0e38f;
    for (int s = lane; s < S; s += 32) mx = fmaxf(mx, sc[r * 1024 + s]);
#pragma unroll
    for (int off = 16; off > 0; off >>= 1) mx = fmaxf(mx, __shfl_xor(mx, off, 32));
    float sum = 0.f;
    for (int s = lane; s < S; s += 32) {
      float e = __expf(sc[r * 1024 + s] - mx);
      sc[r * 1024 + s] = e;
      sum += e;
    }
#pragma unroll
    for (int off = 16; off > 0; off >>= 1) sum += __shfl_xor(sum, off, 32);
    float inv = 1.f / sum;
    for (int s = lane; s < S; s += 32)
      pbb[r * 2048 + s] = (__bf16)(sc[r * 1024 + s] * inv);
  }
  __syncthreads();

  // ---- Phase C: O = P V  (hd = 64 -> 4 column tiles; V transposed: [bh][c][p])
  const __bf16* vbase = vT + (long)bh * HD * T_DIM;
  v8f o[4];
#pragma unroll
  for (int j = 0; j < 4; ++j)
#pragma unroll
    for (int r = 0; r < 8; ++r) o[j][r] = 0.f;

#pragma unroll 1
  for (int ks = 0; ks < S; ks += 32) {
    v16bf pa = load_frag_rowK(pbb, 2048, 0, ks);
    int kp0 = (ks < RT) ? ks : ks + RT * (seg - 1);
#pragma unroll
    for (int j = 0; j < 4; ++j)
      o[j] = wmma_bf16(pa, load_frag_rowK(vbase, T_DIM, j * 16, kp0), o[j]);
  }

#pragma unroll
  for (int j = 0; j < 4; ++j) {
#pragma unroll
    for (int r = 0; r < 8; ++r) {
      int mloc = r + half * 8;
      int c = j * 16 + l15;
      Ao[((long)(p0 + mloc) * B_DIM + b) * E_DIM + h * HD + c] = (__bf16)o[j][r];
    }
  }
}

// ---------------- launcher ----------------

extern "C" void kernel_launch(void* const* d_in, const int* in_sizes, int n_in,
                              void* d_out, int out_size, void* d_ws, size_t ws_size,
                              hipStream_t stream) {
  const float* hs      = (const float*)d_in[0];   // (T, B, E)
  const float* w_qkv   = (const float*)d_in[1];   // (3E, E)
  const float* b_qkv   = (const float*)d_in[2];   // (3E)
  const float* w_out   = (const float*)d_in[3];   // (E, E)
  const float* b_out   = (const float*)d_in[4];   // (E)
  const float* w_rel   = (const float*)d_in[5];   // (512, E)
  const float* b_rel   = (const float*)d_in[6];   // (512)
  const float* smask   = (const float*)d_in[7];   // (rt, rt)
  const float* nmask   = (const float*)d_in[8];   // (NG, 1, rt, 2rt)
  const int*   ib_main = (const int*)d_in[9];     // (B, rt, rt)
  const int*   ib_rel  = (const int*)d_in[10];    // (B, rt, 2rt)
  float* out = (float*)d_out;

  // workspace carve-up (256B aligned)
  char* ws = (char*)d_ws;
  size_t off = 0;
  auto carve = [&](size_t bytes) { char* p = ws + off; off += (bytes + 255) & ~(size_t)255; return p; };
  __bf16* Xb   = (__bf16*)carve((size_t)M_ROWS * E_DIM * 2);       // hidden bf16
  __bf16* Wq   = (__bf16*)carve((size_t)3 * E_DIM * E_DIM * 2);    // in_proj bf16
  __bf16* Wo   = (__bf16*)carve((size_t)E_DIM * E_DIM * 2);        // out_w bf16
  __bf16* Wr   = (__bf16*)carve((size_t)512 * E_DIM * 2);          // rel_w bf16
  __bf16* qbuf = (__bf16*)carve((size_t)64 * T_DIM * HD * 2);      // q  [bh][p][hd]
  __bf16* kbuf = (__bf16*)carve((size_t)64 * T_DIM * HD * 2);      // k  [bh][p][hd]
  __bf16* vTb  = (__bf16*)carve((size_t)64 * HD * T_DIM * 2);      // vT [bh][hd][p]
  float*  vals = (float*)carve((size_t)M_ROWS * 512 * 4);          // rel vals fp32
  __bf16* Ao   = (__bf16*)carve((size_t)M_ROWS * E_DIM * 2);       // attn out bf16

  // 1. conversions to bf16
  {
    long n;
    n = (long)M_ROWS * E_DIM;      cvt_f32_bf16<<<(n + 255) / 256, 256, 0, stream>>>(hs, Xb, n);
    n = (long)3 * E_DIM * E_DIM;   cvt_f32_bf16<<<(n + 255) / 256, 256, 0, stream>>>(w_qkv, Wq, n);
    n = (long)E_DIM * E_DIM;       cvt_f32_bf16<<<(n + 255) / 256, 256, 0, stream>>>(w_out, Wo, n);
    n = (long)512 * E_DIM;         cvt_f32_bf16<<<(n + 255) / 256, 256, 0, stream>>>(w_rel, Wr, n);
  }

  // 2. QKV projection (M=6144, N=3072, K=1024), fused head-layout scatter
  {
    int tiles = (M_ROWS / 32) * (3 * E_DIM / 64);
    gemm_ws<<<tiles / 4, 128, 0, stream>>>(Xb, Wq, b_qkv, nullptr, qbuf, kbuf, vTb,
                                           M_ROWS, 3 * E_DIM, E_DIM, 1);
  }

  // 3. relative-bias values (M=6144, N=512, K=1024)
  {
    int tiles = (M_ROWS / 32) * (512 / 64);
    gemm_ws<<<tiles / 4, 128, 0, stream>>>(Xb, Wr, b_rel, vals, nullptr, nullptr, nullptr,
                                           M_ROWS, 512, E_DIM, 0);
  }

  // 4. attention (main + 2 ngram streams), 4 waves/block, 64KB LDS per wave
  {
    dim3 grid(64, T_DIM / 64);
    size_t shmem = 4 * (16 * 1024 * 4);   // 256KB <= 320KB WGP LDS
    attn_kernel<<<grid, 128, shmem, stream>>>(qbuf, kbuf, vTb, vals, smask, nmask,
                                              ib_main, ib_rel, Ao);
  }

  // 5. output projection (M=6144, N=1024, K=1024) -> fp32 d_out
  {
    int tiles = (M_ROWS / 32) * (E_DIM / 64);
    gemm_ws<<<tiles / 4, 128, 0, stream>>>(Ao, Wo, b_out, out, nullptr, nullptr, nullptr,
                                           M_ROWS, E_DIM, E_DIM, 0);
  }
}
